// LSTMModel_62337155334194
// MI455X (gfx1250) — matrix operand order repaired and checked by reading, not compiled, not run on hardware
//
#include <hip/hip_runtime.h>
#include <hip/hip_bf16.h>

#define EMB    100
#define EMBP   128
#define HID    1024
#define BATCH  256
#define SEQ    512
#define KTOT   (EMBP + HID)     // 1152
#define GCOLS  (4 * HID)        // 4096

typedef __attribute__((ext_vector_type(16))) __bf16        v16bf;
typedef __attribute__((ext_vector_type(8)))  float         v8f;
typedef __attribute__((ext_vector_type(4)))  unsigned int  u32x4;

struct Bf32B { u32x4 lo, hi; };

// A-fragment (16-bit A 16x32): lane holds K [half*16, +16) contiguous
// -> two 16B chunks at p and p+8 elements.
static __device__ __forceinline__ v16bf load_a_frag(const __hip_bfloat16* p) {
    Bf32B q;
    q.lo = *(const u32x4*)(p);
    q.hi = *(const u32x4*)(p + 8);
    return __builtin_bit_cast(v16bf, q);
}
// B-fragment (16-bit B 32x16): lane holds 16 consecutive K of one column
// -> 32 contiguous bytes at p (weights stored transposed, K contiguous).
static __device__ __forceinline__ v16bf load_b_frag(const __hip_bfloat16* p) {
    Bf32B q;
    q.lo = *(const u32x4*)(p);
    q.hi = *(const u32x4*)(p + 8);
    return __builtin_bit_cast(v16bf, q);
}

static __device__ __forceinline__ float sigmoid_fast(float x) {
    return 1.0f / (1.0f + __expf(-x));
}
static __device__ __forceinline__ float tanh_fast(float x) {
    float e = __expf(2.0f * x);
    return (e - 1.0f) / (e + 1.0f);
}

// Build WT[4096][1152] bf16:  WT[n][k] = (k<100) ? W_x{g}[k][col]
//                                        : (k<128) ? 0 : W_h{g}[k-128][col]
// where g = n/1024 (i,f,o,c), col = n%1024.
__global__ void prep_weights(const float* __restrict__ wxi, const float* __restrict__ whi,
                             const float* __restrict__ wxf, const float* __restrict__ whf,
                             const float* __restrict__ wxo, const float* __restrict__ who,
                             const float* __restrict__ wxc, const float* __restrict__ whc,
                             __hip_bfloat16* __restrict__ WT) {
    int idx = blockIdx.x * blockDim.x + threadIdx.x;
    if (idx >= GCOLS * KTOT) return;
    int n = idx / KTOT;
    int k = idx - n * KTOT;
    int g = n >> 10;
    int col = n & (HID - 1);
    const float* wx[4] = {wxi, wxf, wxo, wxc};
    const float* wh[4] = {whi, whf, who, whc};
    float v;
    if (k < EMB)       v = wx[g][k * HID + col];
    else if (k < EMBP) v = 0.0f;
    else               v = wh[g][(k - EMBP) * HID + col];
    WT[idx] = __float2bfloat16(v);
}

// Xbf[s][b][k] = bf16(emb[tokens[b][s]][k]) for k<100 else 0.
__global__ void prep_x(const int* __restrict__ tokens, const float* __restrict__ emb,
                       __hip_bfloat16* __restrict__ Xbf) {
    int idx = blockIdx.x * blockDim.x + threadIdx.x;
    if (idx >= SEQ * BATCH * EMBP) return;
    int s   = idx >> 15;              // / (BATCH*EMBP)
    int rem = idx & (BATCH * EMBP - 1);
    int b   = rem >> 7;
    int k   = rem & (EMBP - 1);
    float v = 0.0f;
    if (k < EMB) {
        int tok = tokens[b * SEQ + s];
        v = emb[(long long)tok * EMB + k];
    }
    Xbf[idx] = __float2bfloat16(v);
}

// H0 -> bf16 buffer, C0 -> fp32 state, concat biases.
__global__ void init_state(const float* __restrict__ H0, const float* __restrict__ C0,
                           const float* __restrict__ bi, const float* __restrict__ bf,
                           const float* __restrict__ bo, const float* __restrict__ bc,
                           __hip_bfloat16* __restrict__ Hb, float* __restrict__ Cst,
                           float* __restrict__ bcat) {
    int idx = blockIdx.x * blockDim.x + threadIdx.x;
    if (idx < BATCH * HID) {
        Hb[idx]  = __float2bfloat16(H0[idx]);
        Cst[idx] = C0[idx];
    }
    if (idx < GCOLS) {
        const float* bs[4] = {bi, bf, bo, bc};
        bcat[idx] = bs[idx >> 10][idx & (HID - 1)];
    }
}

// One LSTM timestep: gates = [X_t | H] @ [W_x ; W_h] + b, fused cell update.
// Block: 128 threads = 4 waves; wave w = col tile ct (16 H-cols).
// Each wave computes TWO row tiles (M=32) x 4 gates = 8 accumulators, so
// every B fragment is reused by 2 WMMAs and every A fragment by 4 WMMAs:
// 12 B128 loads per 8 WMMAs. Block tile: 32 rows x 64 H-cols. Grid: (16, 8).
__global__ __launch_bounds__(128) void lstm_step(
        const __hip_bfloat16* __restrict__ Xbf,    // [SEQ][BATCH][EMBP]
        const __hip_bfloat16* __restrict__ Hprev,  // [BATCH][HID] bf16
        __hip_bfloat16* __restrict__ Hnext,        // [BATCH][HID] bf16
        float* __restrict__ Cst,                   // [BATCH][HID] fp32 (in place)
        const __hip_bfloat16* __restrict__ WT,     // [GCOLS][KTOT] bf16
        const float* __restrict__ bcat,            // [GCOLS]
        float* __restrict__ Hf32,                  // [BATCH][HID] fp32 (last step)
        int t, int last) {
    const int lane = threadIdx.x & 31;
    const int ct   = threadIdx.x >> 5;     // 0..3
    const int half = lane >> 4;
    const int l16  = lane & 15;

    const int row0 = blockIdx.y * 32;                      // block row base
    const int m0   = row0 + l16;                           // row tile 0 lane row
    const int nIn  = blockIdx.x * 64 + ct * 16 + l16;      // H-col for B and C/D

    const __hip_bfloat16* Ax0 = Xbf + ((size_t)t * BATCH + m0) * EMBP + half * 16;
    const __hip_bfloat16* Ax1 = Ax0 + 16 * EMBP;
    const __hip_bfloat16* Ah0 = Hprev + (size_t)m0 * HID + half * 16;
    const __hip_bfloat16* Ah1 = Ah0 + 16 * HID;
    const __hip_bfloat16* Brow0 = WT + (size_t)(0 * HID + nIn) * KTOT + half * 16;
    const __hip_bfloat16* Brow1 = WT + (size_t)(1 * HID + nIn) * KTOT + half * 16;
    const __hip_bfloat16* Brow2 = WT + (size_t)(2 * HID + nIn) * KTOT + half * 16;
    const __hip_bfloat16* Brow3 = WT + (size_t)(3 * HID + nIn) * KTOT + half * 16;

    v8f a0i = {}, a0f = {}, a0o = {}, a0c = {};   // row tile 0
    v8f a1i = {}, a1f = {}, a1o = {}, a1c = {};   // row tile 1

    // Input-projection part of K (k = 0..127, from Xbf)
    #pragma unroll
    for (int kk = 0; kk < EMBP; kk += 32) {
        v16bf a0 = load_a_frag(Ax0 + kk);
        v16bf a1 = load_a_frag(Ax1 + kk);
        v16bf b0 = load_b_frag(Brow0 + kk);
        v16bf b1 = load_b_frag(Brow1 + kk);
        v16bf b2 = load_b_frag(Brow2 + kk);
        v16bf b3 = load_b_frag(Brow3 + kk);
        a0i = __builtin_amdgcn_wmma_f32_16x16x32_bf16(false, a0, false, b0, (short)0, a0i, false, false);
        a1i = __builtin_amdgcn_wmma_f32_16x16x32_bf16(false, a1, false, b0, (short)0, a1i, false, false);
        a0f = __builtin_amdgcn_wmma_f32_16x16x32_bf16(false, a0, false, b1, (short)0, a0f, false, false);
        a1f = __builtin_amdgcn_wmma_f32_16x16x32_bf16(false, a1, false, b1, (short)0, a1f, false, false);
        a0o = __builtin_amdgcn_wmma_f32_16x16x32_bf16(false, a0, false, b2, (short)0, a0o, false, false);
        a1o = __builtin_amdgcn_wmma_f32_16x16x32_bf16(false, a1, false, b2, (short)0, a1o, false, false);
        a0c = __builtin_amdgcn_wmma_f32_16x16x32_bf16(false, a0, false, b3, (short)0, a0c, false, false);
        a1c = __builtin_amdgcn_wmma_f32_16x16x32_bf16(false, a1, false, b3, (short)0, a1c, false, false);
    }
    // Recurrent part of K (k = 128..1151, from Hprev)
    #pragma unroll 2
    for (int kk = 0; kk < HID; kk += 32) {
        v16bf a0 = load_a_frag(Ah0 + kk);
        v16bf a1 = load_a_frag(Ah1 + kk);
        v16bf b0 = load_b_frag(Brow0 + EMBP + kk);
        v16bf b1 = load_b_frag(Brow1 + EMBP + kk);
        v16bf b2 = load_b_frag(Brow2 + EMBP + kk);
        v16bf b3 = load_b_frag(Brow3 + EMBP + kk);
        a0i = __builtin_amdgcn_wmma_f32_16x16x32_bf16(false, a0, false, b0, (short)0, a0i, false, false);
        a1i = __builtin_amdgcn_wmma_f32_16x16x32_bf16(false, a1, false, b0, (short)0, a1i, false, false);
        a0f = __builtin_amdgcn_wmma_f32_16x16x32_bf16(false, a0, false, b1, (short)0, a0f, false, false);
        a1f = __builtin_amdgcn_wmma_f32_16x16x32_bf16(false, a1, false, b1, (short)0, a1f, false, false);
        a0o = __builtin_amdgcn_wmma_f32_16x16x32_bf16(false, a0, false, b2, (short)0, a0o, false, false);
        a1o = __builtin_amdgcn_wmma_f32_16x16x32_bf16(false, a1, false, b2, (short)0, a1o, false, false);
        a0c = __builtin_amdgcn_wmma_f32_16x16x32_bf16(false, a0, false, b3, (short)0, a0c, false, false);
        a1c = __builtin_amdgcn_wmma_f32_16x16x32_bf16(false, a1, false, b3, (short)0, a1c, false, false);
    }

    // Fused cell update. C/D layout: element e -> row = tilebase + 8*half + e,
    // col = nIn (same col across all gate accumulators -> per-lane VALU only).
    const float bi  = bcat[nIn];
    const float bff = bcat[HID + nIn];
    const float bo  = bcat[2 * HID + nIn];
    const float bc  = bcat[3 * HID + nIn];
    #pragma unroll
    for (int rtile = 0; rtile < 2; ++rtile) {
        v8f vi = rtile ? a1i : a0i;
        v8f vf = rtile ? a1f : a0f;
        v8f vo = rtile ? a1o : a0o;
        v8f vc = rtile ? a1c : a0c;
        const int rbase = row0 + rtile * 16 + 8 * half;
        #pragma unroll
        for (int e = 0; e < 8; ++e) {
            size_t idx = (size_t)(rbase + e) * HID + nIn;
            float I  = sigmoid_fast(vi[e] + bi);
            float F  = sigmoid_fast(vf[e] + bff);
            float O  = sigmoid_fast(vo[e] + bo);
            float Ct = tanh_fast(vc[e] + bc);
            float c  = F * Cst[idx] + I * Ct;
            Cst[idx] = c;
            float h  = O * tanh_fast(c);
            Hnext[idx] = __float2bfloat16(h);
            if (last) Hf32[idx] = h;
        }
    }
}

// out[b][j] = sum_k H[b][k] * dense_w[k][j] + dense_b[j]   (256x2, trivial)
__global__ void dense_out(const float* __restrict__ H, const float* __restrict__ dw,
                          const float* __restrict__ db, float* __restrict__ out) {
    int b = threadIdx.x;
    float a0 = db[0], a1 = db[1];
    for (int k = 0; k < HID; ++k) {
        float h = H[(size_t)b * HID + k];
        a0 += h * dw[2 * k];
        a1 += h * dw[2 * k + 1];
    }
    out[2 * b]     = a0;
    out[2 * b + 1] = a1;
}

extern "C" void kernel_launch(void* const* d_in, const int* in_sizes, int n_in,
                              void* d_out, int out_size, void* d_ws, size_t ws_size,
                              hipStream_t stream) {
    const int*   tokens = (const int*)  d_in[0];
    const float* H0     = (const float*)d_in[1];
    const float* C0     = (const float*)d_in[2];
    const float* W_xi   = (const float*)d_in[3];
    const float* W_hi   = (const float*)d_in[4];
    const float* b_i    = (const float*)d_in[5];
    const float* W_xf   = (const float*)d_in[6];
    const float* W_hf   = (const float*)d_in[7];
    const float* b_f    = (const float*)d_in[8];
    const float* W_xo   = (const float*)d_in[9];
    const float* W_ho   = (const float*)d_in[10];
    const float* b_o    = (const float*)d_in[11];
    const float* W_xc   = (const float*)d_in[12];
    const float* W_hc   = (const float*)d_in[13];
    const float* b_c    = (const float*)d_in[14];
    const float* emb    = (const float*)d_in[15];
    const float* dw     = (const float*)d_in[16];
    const float* db     = (const float*)d_in[17];

    size_t off = 0;
    char* base = (char*)d_ws;
    auto alloc = [&](size_t bytes) -> void* {
        void* p = base + off;
        off += (bytes + 255) & ~(size_t)255;
        return p;
    };
    __hip_bfloat16* WT   = (__hip_bfloat16*)alloc((size_t)GCOLS * KTOT * 2);
    __hip_bfloat16* Xbf  = (__hip_bfloat16*)alloc((size_t)SEQ * BATCH * EMBP * 2);
    __hip_bfloat16* Hb0  = (__hip_bfloat16*)alloc((size_t)BATCH * HID * 2);
    __hip_bfloat16* Hb1  = (__hip_bfloat16*)alloc((size_t)BATCH * HID * 2);
    float*          Cst  = (float*)         alloc((size_t)BATCH * HID * 4);
    float*          bcat = (float*)         alloc((size_t)GCOLS * 4);
    float*          Hf32 = (float*)         alloc((size_t)BATCH * HID * 4);

    {
        int n = GCOLS * KTOT;
        prep_weights<<<(n + 255) / 256, 256, 0, stream>>>(W_xi, W_hi, W_xf, W_hf,
                                                          W_xo, W_ho, W_xc, W_hc, WT);
    }
    {
        int n = SEQ * BATCH * EMBP;
        prep_x<<<(n + 255) / 256, 256, 0, stream>>>(tokens, emb, Xbf);
    }
    {
        int n = BATCH * HID;
        init_state<<<(n + 255) / 256, 256, 0, stream>>>(H0, C0, b_i, b_f, b_o, b_c,
                                                        Hb0, Cst, bcat);
    }

    __hip_bfloat16* Hbuf[2] = {Hb0, Hb1};
    dim3 grid(HID / 64, BATCH / 32);   // (16, 8)
    for (int t = 0; t < SEQ; ++t) {
        lstm_step<<<grid, 128, 0, stream>>>(Xbf, Hbuf[t & 1], Hbuf[(t + 1) & 1],
                                            Cst, WT, bcat, Hf32, t, (t == SEQ - 1) ? 1 : 0);
    }
    dense_out<<<1, BATCH, 0, stream>>>(Hf32, dw, db, (float*)d_out);
}
